// TransformerEncoder_14113262534944
// MI455X (gfx1250) — matrix operand order repaired and checked by reading, not compiled, not run on hardware
//
#include <hip/hip_runtime.h>
#include <hip/hip_bf16.h>

typedef __bf16 bf16;
typedef __attribute__((ext_vector_type(8)))  __bf16 bf16x8;
typedef __attribute__((ext_vector_type(16))) __bf16 v16bf;
typedef __attribute__((ext_vector_type(8)))  float  f32x8;
typedef __attribute__((ext_vector_type(4)))  float  f32x4;
typedef __attribute__((ext_vector_type(4)))  unsigned int u32x4;

#define B_  4
#define T_  1024
#define D_  512
#define F_  2048
#define H_  8
#define DH_ 64
#define L_  6
#define M_TOT (B_ * T_)          // 4096 token rows
#define EPS_ 1e-6f
#define NEG_INF_ -1e9f

// ---------------------------------------------------------------------------
// helpers
// ---------------------------------------------------------------------------
__device__ __forceinline__ v16bf ld_frag16(const bf16* p0, const bf16* p1) {
  bf16x8 lo = *(const bf16x8*)p0;
  bf16x8 hi = *(const bf16x8*)p1;
  v16bf r;
#pragma unroll
  for (int i = 0; i < 8; ++i) { r[i] = lo[i]; r[i + 8] = hi[i]; }
  return r;
}

__device__ __forceinline__ f32x8 wmma_bf16(v16bf a, v16bf b, f32x8 c) {
  return __builtin_amdgcn_wmma_f32_16x16x32_bf16(false, a, false, b,
                                                 (short)0, c, false, false);
}

__device__ __forceinline__ unsigned int pack_bf16(float a, float b) {
  union { bf16 h[2]; unsigned int u; } t;
  t.h[0] = (bf16)a; t.h[1] = (bf16)b;
  return t.u;
}

// ---------------------------------------------------------------------------
// weight repack: f32 [K,N] row-major -> bf16 WMMA B-fragment order
// dst[((kt*(N/16)+nt)*32 + lane)*16 + e], K_in_tile = e + 16*(lane/16)
// ---------------------------------------------------------------------------
__global__ void __launch_bounds__(256)
k_pack(const float* __restrict__ w, bf16* __restrict__ o, int K, int N) {
  int idx = blockIdx.x * 256 + threadIdx.x;
  if (idx >= K * N) return;
  int k = idx / N, n = idx % N;
  int kt = k >> 5, kk = k & 31;
  int hl = kk >> 4, e = kk & 15;
  int lane = (n & 15) + 16 * hl;
  int nt = n >> 4;
  size_t dst = ((size_t)(kt * (N >> 4) + nt) * 32 + lane) * 16 + e;
  o[dst] = (bf16)w[(size_t)k * N + n];
}

// ---------------------------------------------------------------------------
// positions + positional embedding + padding bias
// ---------------------------------------------------------------------------
__global__ void __launch_bounds__(256)
k_embed(const float* __restrict__ xin, const float* __restrict__ pos,
        const int* __restrict__ tok, float* __restrict__ x,
        bf16* __restrict__ xb, float* __restrict__ bias) {
  int idx = blockIdx.x * 256 + threadIdx.x;            // over B*T*D
  if (idx >= M_TOT * D_) return;
  int d  = idx & (D_ - 1);
  int bt = idx >> 9;
  int t  = bt & (T_ - 1);
  int tk = tok[bt];
  int p  = (tk != 0) ? (t + 1) : 0;                    // make_positions, pad_idx=0
  float v = xin[idx] + pos[(size_t)p * D_ + d];
  x[idx]  = v;
  xb[idx] = (bf16)v;
  if (d == 0) bias[bt] = (tk == 0) ? NEG_INF_ : 0.0f;
}

// ---------------------------------------------------------------------------
// bf16 WMMA GEMM: C[4096,N] = A[4096,K] @ Wpacked[K,N]
// one wave -> 32x64 strip (2 A-frags x 4 B-frags = 8 WMMA / K-step)
// block = 8 waves = 256 rows x 64 cols
// EPI: 0 f32 out, 1 scale+bf16 (q), 2 bf16 (k), 3 bf16 transposed [b,h,dh,T] (v),
//      4 relu+bf16 (ffn1)
// ---------------------------------------------------------------------------
template <int EPI>
__global__ void __launch_bounds__(256)
k_gemm(const bf16* __restrict__ A, const bf16* __restrict__ Bp,
       float* __restrict__ Cf, bf16* __restrict__ Cb,
       int N, int K, float scale) {
  const int lane = threadIdx.x;
  const int hl = lane >> 4, l16 = lane & 15;
  const int row0 = (blockIdx.y * 8 + threadIdx.y) * 32;
  const int col0 = blockIdx.x * 64;
  const int ntiles = N >> 4;

  f32x8 acc[2][4] = {};
  const bf16* arow0 = A + (size_t)(row0 + l16) * K;
  const bf16* arow1 = A + (size_t)(row0 + 16 + l16) * K;
  const bf16* bcol  = Bp + ((size_t)(col0 >> 4) * 32 + lane) * 16;
  const size_t bks  = (size_t)ntiles * 512;            // elems per K-tile slab

  for (int kb = 0; kb < K; kb += 32) {
    v16bf a0 = ld_frag16(arow0 + kb + 16 * hl, arow0 + kb + 16 * hl + 8);
    v16bf a1 = ld_frag16(arow1 + kb + 16 * hl, arow1 + kb + 16 * hl + 8);
    const bf16* bb = bcol + (size_t)(kb >> 5) * bks;
    if (kb + 32 < K) __builtin_prefetch(bb + bks, 0, 0);
#pragma unroll
    for (int nt = 0; nt < 4; ++nt) {
      v16bf bf = ld_frag16(bb + nt * 512, bb + nt * 512 + 8);
      acc[0][nt] = wmma_bf16(a0, bf, acc[0][nt]);
      acc[1][nt] = wmma_bf16(a1, bf, acc[1][nt]);
    }
  }

#pragma unroll
  for (int rg = 0; rg < 2; ++rg) {
#pragma unroll
    for (int nt = 0; nt < 4; ++nt) {
#pragma unroll
      for (int r = 0; r < 8; ++r) {
        int gm = row0 + rg * 16 + r + 8 * hl;          // C row = r + 8*half
        int gn = col0 + nt * 16 + l16;                 // C col = lane%16
        float v = acc[rg][nt][r];
        if (EPI == 0) {
          Cf[(size_t)gm * N + gn] = v;
        } else if (EPI == 1) {
          Cb[(size_t)gm * N + gn] = (bf16)(v * scale);
        } else if (EPI == 2) {
          Cb[(size_t)gm * N + gn] = (bf16)v;
        } else if (EPI == 3) {                         // v -> vT[b,h,dh,T]
          int b = gm >> 10, t = gm & (T_ - 1);
          int hh = gn >> 6, f = gn & (DH_ - 1);
          Cb[((size_t)(b * H_ + hh) * DH_ + f) * T_ + t] = (bf16)v;
        } else if (EPI == 4) {
          Cb[(size_t)gm * N + gn] = (bf16)fmaxf(v, 0.0f);
        }
      }
    }
  }
}

// ---------------------------------------------------------------------------
// flash attention, transposed-score formulation (no LDS):
//   S^T = K_tile @ Q^T  -> lanes index queries, VGPRs index keys
//   softmax reductions are in-lane VALU trees + one xor-16 shuffle
//   O^T = V^T @ P^T     -> P^T built with 8 packed xor-16 shuffles
// grid (T/64, H, B), block (32,4): one wave per 16-query tile
// q[B,T,D] bf16 (pre-scaled), k[B,T,D] bf16, vT[B,H,DH,T] bf16 -> y[B,T,D] bf16
// ---------------------------------------------------------------------------
__global__ void __launch_bounds__(128)
k_attn(const bf16* __restrict__ q, const bf16* __restrict__ kk,
       const bf16* __restrict__ vT, const float* __restrict__ bias,
       bf16* __restrict__ y) {
  const int lane = threadIdx.x, w = threadIdx.y;
  const int hl = lane >> 4, l16 = lane & 15;
  const int b = blockIdx.z, hh = blockIdx.y;
  const int q0 = (blockIdx.x * 4 + w) * 16;

  // Q^T B-frags: N = query = l16, K-pattern {8*hl.., 16+8*hl..} per kstep*32
  const bf16* qrow = q + ((size_t)(b * T_ + q0 + l16) * D_ + hh * DH_ + 8 * hl);
  v16bf qb0 = ld_frag16(qrow,      qrow + 16);
  v16bf qb1 = ld_frag16(qrow + 32, qrow + 48);

  float m = -1e30f, lsum = 0.0f;                       // per-query running stats
  f32x8 acc[4] = {};                                   // O^T: feat rows x queries
  const float* brow = bias + (size_t)b * T_;

  for (int c = 0; c < T_; c += 32) {
    // K A-frags: M = key = l16 (+tile*16), K-pattern {8*hl.., 16+8*hl..}
    const bf16* k0 = kk + ((size_t)(b * T_ + c + l16) * D_ + hh * DH_);
    const bf16* k1 = k0 + (size_t)16 * D_;
    v16bf ka00 = ld_frag16(k0 + 8 * hl,      k0 + 16 + 8 * hl);
    v16bf ka01 = ld_frag16(k0 + 32 + 8 * hl, k0 + 48 + 8 * hl);
    v16bf ka10 = ld_frag16(k1 + 8 * hl,      k1 + 16 + 8 * hl);
    v16bf ka11 = ld_frag16(k1 + 32 + 8 * hl, k1 + 48 + 8 * hl);
    f32x8 s0 = {}, s1 = {};
    s0 = wmma_bf16(ka00, qb0, s0); s0 = wmma_bf16(ka01, qb1, s0);
    s1 = wmma_bf16(ka10, qb0, s1); s1 = wmma_bf16(ka11, qb1, s1);

    // key-padding bias: s row r <-> key c + tile*16 + 8*hl + r (contiguous)
    f32x4 b0a = *(const f32x4*)(brow + c + 8 * hl);
    f32x4 b0b = *(const f32x4*)(brow + c + 8 * hl + 4);
    f32x4 b1a = *(const f32x4*)(brow + c + 16 + 8 * hl);
    f32x4 b1b = *(const f32x4*)(brow + c + 16 + 8 * hl + 4);
    float t0[8], t1[8];
#pragma unroll
    for (int r = 0; r < 8; ++r) {
      t0[r] = s0[r] + (r < 4 ? b0a[r] : b0b[r - 4]);
      t1[r] = s1[r] + (r < 4 ? b1a[r] : b1b[r - 4]);
    }

    // chunk max: in-lane tree over 16 keys, then combine key-halves (xor 16)
    float m0 = fmaxf(fmaxf(fmaxf(t0[0], t0[1]), fmaxf(t0[2], t0[3])),
                     fmaxf(fmaxf(t0[4], t0[5]), fmaxf(t0[6], t0[7])));
    float m1 = fmaxf(fmaxf(fmaxf(t1[0], t1[1]), fmaxf(t1[2], t1[3])),
                     fmaxf(fmaxf(t1[4], t1[5]), fmaxf(t1[6], t1[7])));
    float cm = fmaxf(m0, m1);
    cm = fmaxf(cm, __shfl_xor(cm, 16, 32));
    float nm = fmaxf(m, cm);
    float sc = __expf(m - nm);

    float p0[8], p1[8];
#pragma unroll
    for (int r = 0; r < 8; ++r) {
      p0[r] = __expf(t0[r] - nm);
      p1[r] = __expf(t1[r] - nm);
    }
    float rs = ((p0[0] + p0[1]) + (p0[2] + p0[3])) +
               ((p0[4] + p0[5]) + (p0[6] + p0[7])) +
               ((p1[0] + p1[1]) + (p1[2] + p1[3])) +
               ((p1[4] + p1[5]) + (p1[6] + p1[7]));
    rs += __shfl_xor(rs, 16, 32);
    lsum = lsum * sc + rs;
    m = nm;
#pragma unroll
    for (int nt = 0; nt < 4; ++nt)
#pragma unroll
      for (int r = 0; r < 8; ++r) acc[nt][r] *= sc;

    // build P^T B-frag: lane = query, element e = key e + 16*hl
    // own half holds keys {8*hl..8*hl+7} (tile0) / {16+8*hl..} (tile1);
    // exchange packed bf16 pairs across halves (xor 16)
    unsigned int own0[4], own1[4], oth0[4], oth1[4];
#pragma unroll
    for (int j = 0; j < 4; ++j) {
      own0[j] = pack_bf16(p0[2 * j], p0[2 * j + 1]);
      own1[j] = pack_bf16(p1[2 * j], p1[2 * j + 1]);
    }
#pragma unroll
    for (int j = 0; j < 4; ++j) {
      oth0[j] = __shfl_xor(own0[j], 16, 32);
      oth1[j] = __shfl_xor(own1[j], 16, 32);
    }
    union { unsigned int u[8]; v16bf v; } pb;
#pragma unroll
    for (int j = 0; j < 4; ++j) {
      pb.u[j]     = hl ? oth1[j] : own0[j];            // keys (0..7)+16*hl
      pb.u[4 + j] = hl ? own1[j] : oth0[j];            // keys (8..15)+16*hl
    }

    // O^T += V^T_tile @ P^T  (A: M=feature=l16, K = e + 16*hl)
#pragma unroll
    for (int nt = 0; nt < 4; ++nt) {
      const bf16* vrow =
          vT + (((size_t)(b * H_ + hh) * DH_ + nt * 16 + l16) * T_ + c);
      v16bf va = ld_frag16(vrow + 16 * hl, vrow + 16 * hl + 8);
      acc[nt] = wmma_bf16(va, pb.v, acc[nt]);
    }
  }

  // normalize + packed contiguous store: lane=token row, feats r contiguous
  float inv = 1.0f / lsum;
  bf16* yrow = y + ((size_t)(b * T_ + q0 + l16) * D_ + hh * DH_ + 8 * hl);
#pragma unroll
  for (int nt = 0; nt < 4; ++nt) {
    u32x4 o;
#pragma unroll
    for (int j = 0; j < 4; ++j)
      o[j] = pack_bf16(acc[nt][2 * j] * inv, acc[nt][2 * j + 1] * inv);
    *(u32x4*)(void*)(yrow + nt * 16) = o;
  }
}

// ---------------------------------------------------------------------------
// residual + unbiased LayerNorm (divide by sqrt(var)+eps), one row per block
// ---------------------------------------------------------------------------
__global__ void __launch_bounds__(256)
k_resln(const float* __restrict__ xin, const float* __restrict__ yin,
        const float* __restrict__ g, const float* __restrict__ bb,
        float* __restrict__ xout, bf16* __restrict__ xbout) {
  __shared__ float red[256];
  const int row = blockIdx.x, tid = threadIdx.x;
  const size_t base = (size_t)row * D_;
  float v0 = xin[base + tid]       + yin[base + tid];
  float v1 = xin[base + tid + 256] + yin[base + tid + 256];
  red[tid] = v0 + v1;
  __syncthreads();
  for (int o = 128; o > 0; o >>= 1) {
    if (tid < o) red[tid] += red[tid + o];
    __syncthreads();
  }
  float mean = red[0] * (1.0f / D_);
  __syncthreads();
  float d0 = v0 - mean, d1 = v1 - mean;
  red[tid] = d0 * d0 + d1 * d1;
  __syncthreads();
  for (int o = 128; o > 0; o >>= 1) {
    if (tid < o) red[tid] += red[tid + o];
    __syncthreads();
  }
  float var   = red[0] * (1.0f / (D_ - 1));            // unbiased (Bessel)
  float denom = sqrtf(var) + EPS_;
  float o0 = g[tid]       * d0 / denom + bb[tid];
  float o1 = g[tid + 256] * d1 / denom + bb[tid + 256];
  xout[base + tid]        = o0;
  xout[base + tid + 256]  = o1;
  xbout[base + tid]       = (bf16)o0;
  xbout[base + tid + 256] = (bf16)o1;
}

// ---------------------------------------------------------------------------
// host orchestration
// ---------------------------------------------------------------------------
extern "C" void kernel_launch(void* const* d_in, const int* in_sizes, int n_in,
                              void* d_out, int out_size, void* d_ws, size_t ws_size,
                              hipStream_t stream) {
  (void)in_sizes; (void)n_in; (void)out_size; (void)ws_size;
  const float* x_in = (const float*)d_in[0];
  const float* pos  = (const float*)d_in[1];
  const float* wq   = (const float*)d_in[2];
  const float* wk   = (const float*)d_in[3];
  const float* wv   = (const float*)d_in[4];
  const float* wo   = (const float*)d_in[5];
  const float* w1   = (const float*)d_in[6];
  const float* w2   = (const float*)d_in[7];
  const float* g1   = (const float*)d_in[8];
  const float* b1   = (const float*)d_in[9];
  const float* g2   = (const float*)d_in[10];
  const float* b2   = (const float*)d_in[11];
  const int*   tok  = (const int*)d_in[12];
  float* out = (float*)d_out;

  char* ws = (char*)d_ws;
  size_t off = 0;
  auto alloc = [&](size_t bytes) -> void* {
    off = (off + 255) & ~(size_t)255;
    void* p = ws + off;
    off += bytes;
    return p;
  };

  const size_t DD = (size_t)D_ * D_;                   // 262144
  const size_t DF = (size_t)D_ * F_;                   // 1048576
  const size_t LAYER = 4 * DD + 2 * DF;                // packed elems/layer

  bf16*  wpack = (bf16*)alloc(L_ * LAYER * sizeof(bf16));
  float* xbuf  = (float*)alloc((size_t)M_TOT * D_ * sizeof(float));
  bf16*  xb    = (bf16*)alloc((size_t)M_TOT * D_ * sizeof(bf16));
  bf16*  qb    = (bf16*)alloc((size_t)M_TOT * D_ * sizeof(bf16));
  bf16*  kb    = (bf16*)alloc((size_t)M_TOT * D_ * sizeof(bf16));
  bf16*  vTb   = (bf16*)alloc((size_t)M_TOT * D_ * sizeof(bf16));
  bf16*  yb    = (bf16*)alloc((size_t)M_TOT * D_ * sizeof(bf16));
  bf16*  f1b   = (bf16*)alloc((size_t)M_TOT * F_ * sizeof(bf16));
  float* t1    = (float*)alloc((size_t)M_TOT * D_ * sizeof(float));
  float* biasb = (float*)alloc((size_t)M_TOT * sizeof(float));

  // 1) repack all weights to bf16 fragment order (deterministic each call)
  for (int l = 0; l < L_; ++l) {
    bf16* lw = wpack + (size_t)l * LAYER;
    k_pack<<<(int)(DD / 256), 256, 0, stream>>>(wq + l * DD, lw,           D_, D_);
    k_pack<<<(int)(DD / 256), 256, 0, stream>>>(wk + l * DD, lw + DD,      D_, D_);
    k_pack<<<(int)(DD / 256), 256, 0, stream>>>(wv + l * DD, lw + 2 * DD,  D_, D_);
    k_pack<<<(int)(DD / 256), 256, 0, stream>>>(wo + l * DD, lw + 3 * DD,  D_, D_);
    k_pack<<<(int)(DF / 256), 256, 0, stream>>>(w1 + l * DF, lw + 4 * DD,  D_, F_);
    k_pack<<<(int)(DF / 256), 256, 0, stream>>>(w2 + l * DF, lw + 4 * DD + DF, F_, D_);
  }

  // 2) embedding + bias
  k_embed<<<(M_TOT * D_) / 256, 256, 0, stream>>>(x_in, pos, tok, xbuf, xb, biasb);

  // 3) layers
  dim3 blk(32, 8);
  dim3 gdD(D_ / 64, M_TOT / 256);                      // (8, 16)
  dim3 gdF(F_ / 64, M_TOT / 256);                      // (32, 16)
  const float qscale = 0.125f;                         // dh^-0.5 = 1/8

  for (int l = 0; l < L_; ++l) {
    bf16* lw = wpack + (size_t)l * LAYER;
    k_gemm<1><<<gdD, blk, 0, stream>>>(xb, lw,          nullptr, qb,  D_, D_, qscale);
    k_gemm<2><<<gdD, blk, 0, stream>>>(xb, lw + DD,     nullptr, kb,  D_, D_, 1.0f);
    k_gemm<3><<<gdD, blk, 0, stream>>>(xb, lw + 2 * DD, nullptr, vTb, D_, D_, 1.0f);
    k_attn<<<dim3(T_ / 64, H_, B_), dim3(32, 4), 0, stream>>>(qb, kb, vTb, biasb, yb);
    k_gemm<0><<<gdD, blk, 0, stream>>>(yb, lw + 3 * DD, t1, nullptr, D_, D_, 1.0f);
    k_resln<<<M_TOT, 256, 0, stream>>>(xbuf, t1, g1 + l * D_, b1 + l * D_, xbuf, xb);
    k_gemm<4><<<gdF, blk, 0, stream>>>(xb, lw + 4 * DD, nullptr, f1b, F_, D_, 1.0f);
    k_gemm<0><<<gdD, blk, 0, stream>>>(f1b, lw + 4 * DD + DF, t1, nullptr, D_, F_, 1.0f);
    k_resln<<<M_TOT, 256, 0, stream>>>(xbuf, t1, g2 + l * D_, b2 + l * D_, xbuf, xb);
  }

  // 4) final output (f32)
  hipMemcpyAsync(out, xbuf, (size_t)M_TOT * D_ * sizeof(float),
                 hipMemcpyDeviceToDevice, stream);
}
